// BBOX_NET_15247133901557
// MI455X (gfx1250) — matrix operand order repaired and checked by reading, not compiled, not run on hardware
//
#include <hip/hip_runtime.h>
#include <hip/hip_bf16.h>

typedef __attribute__((ext_vector_type(16))) _Float16 v16h;
typedef __attribute__((ext_vector_type(8)))  _Float16 v8h;
typedef __attribute__((ext_vector_type(8)))  float    v8f;

#define BATCH   2048
#define CDIM    128
#define OBJ     10
#define LRELU_S 0.2f
#define BNEPS   1e-5f

__device__ __forceinline__ float lrelu(float v) { return v >= 0.0f ? v : LRELU_S * v; }

// separable in-bounds bilinear coverage weight for one axis (n = 16)
__device__ __forceinline__ float cov1d(float g) {
    float ix = (g + 1.0f) * 8.0f - 0.5f;     // ((g+1)*16 - 1) * 0.5
    float x0 = floorf(ix);
    float fx = ix - x0;
    float w = 0.0f;
    if (x0 >= 0.0f  && x0 <= 15.0f) w += 1.0f - fx;
    if (x0 >= -1.0f && x0 <= 14.0f) w += fx;
    return w;
}

// ---------------------------------------------------------------------------
// repack conv weights (OC, IC, 3, 3) f32 -> f16 transposed GEMM matrix
// wt[k][oc] with k = tap*IC + ic (tap-major => a 32-wide K tile is one tap).
// ---------------------------------------------------------------------------
__global__ void repack_w(const float* __restrict__ w, _Float16* __restrict__ wt,
                         int OC, int IC) {
    int n = OC * IC * 9;
    int i = blockIdx.x * blockDim.x + threadIdx.x;
    if (i >= n) return;
    int oc  = i / (IC * 9);
    int r   = i - oc * IC * 9;
    int ic  = r / 9;
    int tap = r - ic * 9;
    wt[(tap * IC + ic) * OC + oc] = (_Float16)w[i];
}

__global__ void zero_f32(float* __restrict__ p, int n) {
    int i = blockIdx.x * blockDim.x + threadIdx.x;
    if (i < n) p[i] = 0.0f;
}

// ---------------------------------------------------------------------------
// Fused: coverage + layout einsum (K=10, VALU) + conv1 WMMA.
// GEMM: A = layout patches (M=64 positions), B = w1t (N=64 oc), K=1152.
// Layout lives only in LDS as a zero-padded 18x18 x 128ch tile (81 KB):
// all fragment loads are branch-free aligned b128.
// One workgroup (256 thr = 8 waves) per image; wave = (ptile-pair, octile),
// one B fragment feeds two WMMAs.
// ---------------------------------------------------------------------------
__global__ __launch_bounds__(256) void layout_conv1(
    const float*    __restrict__ labels,   // (B, O, C)
    const float*    __restrict__ theta,    // (B, O, 2, 3)
    const _Float16* __restrict__ w1t,      // (1152, 64)
    _Float16*       __restrict__ h1t)      // (B, 64 pos, 64 oc)
{
    __shared__ __align__(16) _Float16 lay[324 * 128];   // padded 18x18, ch-minor
    const int b = blockIdx.x;
    const int t = threadIdx.x;

    // ---- zero the padded tile (borders stay zero) ----
    {
        uint4* z = (uint4*)lay;
        const uint4 Z4 = make_uint4(0u, 0u, 0u, 0u);
        for (int i = t; i < 324 * 128 / 8; i += 256) z[i] = Z4;  // 5184 uint4
    }
    __syncthreads();

    // ---- phase 1: layout[p][c] = sum_o labels[b,o,c] * cov[b,o,p] ----
    {
        const int c  = t >> 1;
        const int p0 = (t & 1) * 128;
        float lab[OBJ];
        float th[OBJ][6];
#pragma unroll
        for (int o = 0; o < OBJ; ++o) {
            lab[o] = labels[(b * OBJ + o) * CDIM + c];
#pragma unroll
            for (int j = 0; j < 6; ++j) th[o][j] = theta[b * OBJ * 6 + o * 6 + j];
        }
        for (int p = p0; p < p0 + 128; ++p) {
            const int ry = p >> 4, rx = p & 15;
            const float xj = (2.0f * rx + 1.0f) * 0.0625f - 1.0f;
            const float xi = (2.0f * ry + 1.0f) * 0.0625f - 1.0f;
            float acc = 0.0f;
#pragma unroll
            for (int o = 0; o < OBJ; ++o) {
                float gx = th[o][0] * xj + th[o][1] * xi + th[o][2];
                float gy = th[o][3] * xj + th[o][4] * xi + th[o][5];
                acc += lab[o] * (cov1d(gx) * cov1d(gy));
            }
            lay[((ry + 1) * 18 + rx + 1) * 128 + c] = (_Float16)acc;
        }
    }
    __syncthreads();

    // ---- phase 2: conv1 WMMA ----
    const int wave   = t >> 5;
    const int lane   = t & 31;
    const int laneHi = (lane >= 16) ? 1 : 0;
    const int octile = wave & 3;             // N tile (output channels)
    const int pt0    = (wave >> 2) * 2;      // first of two M tiles (positions)
    const int m      = lane & 15;

    v8f acc0 = {}, acc1 = {};
    for (int ks = 0; ks < 36; ++ks) {
        const int kb  = ks * 32;
        const int tap = kb >> 7;             // k = tap*128 + ic
        const int ky  = tap / 3, kx = tap - ky * 3;
        const int icb = (kb & 127) + (laneHi ? 8 : 0);

        // B fragment: lane = oc column lane&15, element s = K row kb+8*laneHi+(s&7)+16*(s>>3)
        const _Float16* wp = w1t + (size_t)kb * 64 + octile * 16 + (lane & 15);
        union { v16h v; v8h h[2]; } bf;
        for (int s = 0; s < 16; ++s)
            bf.v[s] = wp[(size_t)(laneHi * 8 + (s & 7) + (s >> 3) * 16) * 64];

        // A fragments: 8 contiguous channels at one padded pixel
        const int p0  = pt0 * 16 + m;
        const int iy0 = 2 * (p0 >> 3) + ky;  // (2*oy - 1 + ky) + 1 pad
        const int ix0 = 2 * (p0 & 7)  + kx;
        const _Float16* ap0 = lay + (iy0 * 18 + ix0) * 128 + icb;
        union { v16h v; v8h h[2]; } a0;
        a0.h[0] = *(const v8h*)(ap0);
        a0.h[1] = *(const v8h*)(ap0 + 16);

        const int p1  = p0 + 16;
        const int iy1 = 2 * (p1 >> 3) + ky;
        const int ix1 = 2 * (p1 & 7)  + kx;
        const _Float16* ap1 = lay + (iy1 * 18 + ix1) * 128 + icb;
        union { v16h v; v8h h[2]; } a1;
        a1.h[0] = *(const v8h*)(ap1);
        a1.h[1] = *(const v8h*)(ap1 + 16);

        acc0 = __builtin_amdgcn_wmma_f32_16x16x32_f16(
            false, a0.v, false, bf.v, (short)0, acc0, false, false);
        acc1 = __builtin_amdgcn_wmma_f32_16x16x32_f16(
            false, a1.v, false, bf.v, (short)0, acc1, false, false);
    }

    // D element (r, lane): M = r + 8*laneHi (position), N = lane&15 (oc)
    const int oc = octile * 16 + (lane & 15);
#pragma unroll
    for (int r = 0; r < 8; ++r) {
        const int pA = pt0 * 16 + r + (laneHi ? 8 : 0);
        h1t[((size_t)b * 64 + pA) * 64 + oc]        = (_Float16)lrelu(acc0[r]);
        h1t[((size_t)b * 64 + pA + 16) * 64 + oc]   = (_Float16)lrelu(acc1[r]);
    }
}

// ---------------------------------------------------------------------------
// conv2: A = h1 patches (M=16 positions), B = w2t (N=32 oc, 2 tiles), K=576.
// 2 waves/image; input staged as zero-padded 10x10 x 64ch LDS tile.
// Emits pre-BN f32 [b][pos][ch] + per-channel sum/sum^2 atomics.
// ---------------------------------------------------------------------------
__global__ __launch_bounds__(64) void conv2_wmma(
    const _Float16* __restrict__ h1t, const _Float16* __restrict__ w2t,
    float* __restrict__ h2pre, float* __restrict__ sum2, float* __restrict__ sq2)
{
    __shared__ __align__(16) _Float16 xin[100 * 64];   // padded 10x10, ch-minor
    const int b = blockIdx.x;
    const int t = threadIdx.x;
    {
        uint4* z = (uint4*)xin;
        const uint4 Z4 = make_uint4(0u, 0u, 0u, 0u);
        for (int i = t; i < 800; i += 64) z[i] = Z4;
    }
    __syncthreads();
    for (int cid = t; cid < 512; cid += 64) {          // 8-ch chunks
        const int p = cid >> 3, part = cid & 7;
        const v8h s = *(const v8h*)(h1t + ((size_t)b * 64 + p) * 64 + part * 8);
        const int pix = ((p >> 3) + 1) * 10 + (p & 7) + 1;
        *(v8h*)(xin + pix * 64 + part * 8) = s;
    }
    __syncthreads();

    const int ntile  = t >> 5;     // oc tile
    const int lane   = t & 31;
    const int laneHi = (lane >= 16) ? 1 : 0;
    const int m      = lane & 15;

    v8f acc = {};
    for (int ks = 0; ks < 18; ++ks) {
        const int kb  = ks * 32;
        const int tap = kb >> 6;               // k = tap*64 + ic
        const int ky  = tap / 3, kx = tap - ky * 3;
        const int icb = (kb & 63) + (laneHi ? 8 : 0);

        const _Float16* wp = w2t + (size_t)kb * 32 + ntile * 16 + (lane & 15);
        union { v16h v; v8h h[2]; } bf;
        for (int s = 0; s < 16; ++s)
            bf.v[s] = wp[(size_t)(laneHi * 8 + (s & 7) + (s >> 3) * 16) * 32];

        const int iy = 2 * (m >> 2) + ky;      // padded
        const int ix = 2 * (m & 3)  + kx;
        const _Float16* ap = xin + (iy * 10 + ix) * 64 + icb;
        union { v16h v; v8h h[2]; } a;
        a.h[0] = *(const v8h*)(ap);
        a.h[1] = *(const v8h*)(ap + 16);

        acc = __builtin_amdgcn_wmma_f32_16x16x32_f16(
            false, a.v, false, bf.v, (short)0, acc, false, false);
    }

    const int ch = ntile * 16 + (lane & 15);
#pragma unroll
    for (int r = 0; r < 8; ++r) {
        const int pos = r + (laneHi ? 8 : 0);
        float v = acc[r];
        h2pre[(size_t)b * 512 + pos * 32 + ch] = v;
        atomicAdd(&sum2[ch], v);
        atomicAdd(&sq2[ch], v * v);
    }
}

// ---------------------------------------------------------------------------
// conv3: 4 images packed per M tile (M = 4 img x 4 pos), N=16 oc, K=288.
// 1 wave per 4 images; zero-padded 6x6 x 32ch tile per image.
// ---------------------------------------------------------------------------
__global__ __launch_bounds__(32) void conv3_wmma(
    const _Float16* __restrict__ h2, const _Float16* __restrict__ w3t,
    float* __restrict__ h3pre, float* __restrict__ sum3, float* __restrict__ sq3)
{
    __shared__ __align__(16) _Float16 xin[4 * 36 * 32];   // 4 padded 6x6 tiles
    const int b0 = blockIdx.x * 4;
    const int t  = threadIdx.x;
    {
        uint4* z = (uint4*)xin;
        const uint4 Z4 = make_uint4(0u, 0u, 0u, 0u);
        for (int i = t; i < 576; i += 32) z[i] = Z4;
    }
    __syncthreads();
    for (int cid = t; cid < 256; cid += 32) {             // 8-ch chunks
        const int img = cid >> 6, rem = cid & 63;
        const int n = rem >> 2, part = rem & 3;
        const v8h s = *(const v8h*)(h2 + (size_t)(b0 + img) * 512 + n * 32 + part * 8);
        const int pix = ((n >> 2) + 1) * 6 + (n & 3) + 1;
        *(v8h*)(xin + img * 1152 + pix * 32 + part * 8) = s;
    }
    __syncthreads();

    const int lane   = t;
    const int laneHi = (lane >= 16) ? 1 : 0;
    const int m      = lane & 15;

    v8f acc = {};
    for (int tap = 0; tap < 9; ++tap) {                   // k = tap*32 + ic
        const int ky  = tap / 3, kx = tap - ky * 3;
        const int icb = (laneHi ? 8 : 0);

        const _Float16* wp = w3t + (size_t)(tap * 32) * 16 + (lane & 15);
        union { v16h v; v8h h[2]; } bf;
        for (int s = 0; s < 16; ++s)
            bf.v[s] = wp[(size_t)(laneHi * 8 + (s & 7) + (s >> 3) * 16) * 16];

        const int img = m >> 2, pp = m & 3;
        const int iy = 2 * (pp >> 1) + ky;                // padded
        const int ix = 2 * (pp & 1)  + kx;
        const _Float16* ap = xin + img * 1152 + (iy * 6 + ix) * 32 + icb;
        union { v16h v; v8h h[2]; } a;
        a.h[0] = *(const v8h*)(ap);
        a.h[1] = *(const v8h*)(ap + 16);

        acc = __builtin_amdgcn_wmma_f32_16x16x32_f16(
            false, a.v, false, bf.v, (short)0, acc, false, false);
    }

    const int ch = lane & 15;                             // N = oc
#pragma unroll
    for (int r = 0; r < 8; ++r) {
        const int mm  = r + (laneHi ? 8 : 0);
        const int img = mm >> 2, p = mm & 3;
        float v = acc[r];
        h3pre[(size_t)(b0 + img) * 64 + ch * 4 + p] = v;
        atomicAdd(&sum3[ch], v);
        atomicAdd(&sq3[ch], v * v);
    }
}

// ---------------------------------------------------------------------------
// BN finalize + apply
// ---------------------------------------------------------------------------
__global__ void bn_finalize(const float* __restrict__ sum, const float* __restrict__ sq,
                            const float* __restrict__ gamma, const float* __restrict__ beta,
                            float* __restrict__ scale, float* __restrict__ shift,
                            int C, float invN)
{
    int c = threadIdx.x;
    if (c >= C) return;
    float mean = sum[c] * invN;
    float var  = sq[c] * invN - mean * mean;
    float s    = gamma[c] * rsqrtf(var + BNEPS);
    scale[c] = s;
    shift[c] = beta[c] - mean * s;
}

__global__ void bn_apply2(const float* __restrict__ h2pre, const float* __restrict__ scale,
                          const float* __restrict__ shift, _Float16* __restrict__ h2, int n)
{
    int i = blockIdx.x * blockDim.x + threadIdx.x;
    if (i >= n) return;
    int ch = i & 31;                                  // [b][pos][ch] layout
    h2[i] = (_Float16)lrelu(h2pre[i] * scale[ch] + shift[ch]);
}

__global__ void bn_apply3(const float* __restrict__ h3pre, const float* __restrict__ scale,
                          const float* __restrict__ shift, float* __restrict__ out, int n)
{
    int i = blockIdx.x * blockDim.x + threadIdx.x;
    if (i >= n) return;
    int ch = (i >> 2) & 15;                           // [b][ch][2x2] layout
    out[i] = lrelu(h3pre[i] * scale[ch] + shift[ch]);
}

// ---------------------------------------------------------------------------
// workspace layout (bytes)
// ---------------------------------------------------------------------------
static constexpr size_t O_W1T = 0;                                         // 1152x64 f16
static constexpr size_t O_W2T = O_W1T + (size_t)1152 * 64 * 2;             // 576x32  f16
static constexpr size_t O_W3T = O_W2T + (size_t)576 * 32 * 2;              // 288x16  f16
static constexpr size_t O_H1  = O_W3T + (size_t)288 * 16 * 2;              // B*64*64 f16
static constexpr size_t O_H2P = O_H1  + (size_t)BATCH * 64 * 64 * 2;       // B*512   f32
static constexpr size_t O_H2  = O_H2P + (size_t)BATCH * 512 * 4;           // B*512   f16
static constexpr size_t O_H3P = O_H2  + (size_t)BATCH * 512 * 2;           // B*64    f32
static constexpr size_t O_ST  = O_H3P + (size_t)BATCH * 64 * 4;            // 192 f32

extern "C" void kernel_launch(void* const* d_in, const int* in_sizes, int n_in,
                              void* d_out, int out_size, void* d_ws, size_t ws_size,
                              hipStream_t stream) {
    const float* labels = (const float*)d_in[0];   // (2048, 10, 128)
    const float* theta  = (const float*)d_in[1];   // (2048, 10, 2, 3)
    const float* w1     = (const float*)d_in[2];   // (64, 128, 3, 3)
    const float* w2     = (const float*)d_in[3];   // (32, 64, 3, 3)
    const float* gamma2 = (const float*)d_in[4];
    const float* beta2  = (const float*)d_in[5];
    const float* w3     = (const float*)d_in[6];   // (16, 32, 3, 3)
    const float* gamma3 = (const float*)d_in[7];
    const float* beta3  = (const float*)d_in[8];
    // d_in[9] = max_objects == 10 == full O dimension; folded into constants.

    char* ws = (char*)d_ws;
    _Float16* w1t   = (_Float16*)(ws + O_W1T);
    _Float16* w2t   = (_Float16*)(ws + O_W2T);
    _Float16* w3t   = (_Float16*)(ws + O_W3T);
    _Float16* h1t   = (_Float16*)(ws + O_H1);
    float*    h2pre = (float*)   (ws + O_H2P);
    _Float16* h2    = (_Float16*)(ws + O_H2);
    float*    h3pre = (float*)   (ws + O_H3P);
    float*    st    = (float*)   (ws + O_ST);
    float* sum2 = st;        float* sq2 = st + 32;
    float* sum3 = st + 64;   float* sq3 = st + 80;
    float* sc2  = st + 96;   float* sh2 = st + 128;
    float* sc3  = st + 160;  float* sh3 = st + 176;

    float* out = (float*)d_out;                    // (2048, 64) f32

    zero_f32<<<1, 256, 0, stream>>>(st, 192);
    repack_w<<<(64 * 128 * 9 + 255) / 256, 256, 0, stream>>>(w1, w1t, 64, 128);
    repack_w<<<(32 * 64 * 9 + 255) / 256, 256, 0, stream>>>(w2, w2t, 32, 64);
    repack_w<<<(16 * 32 * 9 + 255) / 256, 256, 0, stream>>>(w3, w3t, 16, 32);

    layout_conv1<<<BATCH, 256, 0, stream>>>(labels, theta, w1t, h1t);
    conv2_wmma<<<BATCH, 64, 0, stream>>>(h1t, w2t, h2pre, sum2, sq2);
    bn_finalize<<<1, 32, 0, stream>>>(sum2, sq2, gamma2, beta2, sc2, sh2, 32,
                                      1.0f / (float)(BATCH * 16));
    bn_apply2<<<(BATCH * 512 + 255) / 256, 256, 0, stream>>>(h2pre, sc2, sh2, h2, BATCH * 512);
    conv3_wmma<<<BATCH / 4, 32, 0, stream>>>(h2, w3t, h3pre, sum3, sq3);
    bn_finalize<<<1, 32, 0, stream>>>(sum3, sq3, gamma3, beta3, sc3, sh3, 16,
                                      1.0f / (float)(BATCH * 4));
    bn_apply3<<<(BATCH * 64 + 255) / 256, 256, 0, stream>>>(h3pre, sc3, sh3, out, BATCH * 64);
}
